// SelfAttentionLayer_23871428231473
// MI455X (gfx1250) — matrix operand order repaired and checked by reading, not compiled, not run on hardware
//
#include <hip/hip_runtime.h>
#include <hip/hip_bf16.h>

// ---------------------------------------------------------------------------
// Self-attention (S=4096, D=2048), fp32 in/out, bf16 WMMA internally.
//   Q = X@Wq^T + bq ; K = X@Wk^T + bk ; V = X@Wv^T + bv
//   QK = Q@K^T / sqrt(D) ; P = tril(softmax_fullrow(QK)) ; out = P@V
// All GEMMs are "NT" (both operands row-major, contiguous K axis) so A/B
// WMMA fragments are plain contiguous global loads (no transpose needed).
// V is stored transposed (Vt[D][S]) so P@V is also NT.
// Per-wave output tile: 64x64 (4x4 fragments, 16 WMMAs per 32-wide K step)
// to maximize register-level operand reuse (16 B/lane of loads per WMMA).
// ---------------------------------------------------------------------------

#define S_DIM 4096
#define D_DIM 2048

typedef __bf16 bf16_t;
typedef __attribute__((ext_vector_type(16))) __bf16 bf16x16;
typedef __attribute__((ext_vector_type(8)))  __bf16 bf16x8;
typedef __attribute__((ext_vector_type(4)))  __bf16 bf16x4;
typedef __attribute__((ext_vector_type(8)))  float  f32x8;

// ---------------------------------------------------------------------------
// fp32 -> bf16 conversion, 4 elements / thread
// ---------------------------------------------------------------------------
__global__ void cvt_f32_to_bf16(const float* __restrict__ in,
                                bf16_t* __restrict__ out, int n4) {
  int i = blockIdx.x * blockDim.x + threadIdx.x;
  if (i >= n4) return;
  float4 v = reinterpret_cast<const float4*>(in)[i];
  bf16x4 o;
  o.x = (bf16_t)v.x; o.y = (bf16_t)v.y; o.z = (bf16_t)v.z; o.w = (bf16_t)v.w;
  reinterpret_cast<bf16x4*>(out)[i] = o;
}

// ---------------------------------------------------------------------------
// Per-wave 64x64 output tile of A[M,K] @ B[N,K]^T, bf16 inputs, f32 accum.
// CDNA5 16x16x32 bf16 fragment layout (wave32):
//   A frag: lane r(0..15)/half h: row = row0+r, elems 0..7 = K[k0+8h .. +7],
//           elems 8..15 = K[k0+16+8h .. +7]        (two 16B contiguous loads)
//   B frag: lane r/half h: col = col0+r, elems 0..7 = K[k0+8h .. +7],
//           elems 8..15 = K[k0+16+8h .. +7]        (two 16B contiguous loads)
//   C/D:    vgpr j, lane r/half h -> (row = 8h + j, col = r)
// ---------------------------------------------------------------------------
union Frag16 { bf16x16 v; bf16x8 h[2]; };

__device__ __forceinline__ void wmma_tile_64x64(const bf16_t* __restrict__ A,
                                                const bf16_t* __restrict__ B,
                                                int K, int row0, int col0,
                                                f32x8 acc[4][4]) {
  const int lane = threadIdx.x & 31;
  const int half = lane >> 4;
  const int r    = lane & 15;

  const bf16_t* arow[4];
  const bf16_t* brow[4];
#pragma unroll
  for (int i = 0; i < 4; ++i) {
    arow[i] = A + (size_t)(row0 + i * 16 + r) * K;
    brow[i] = B + (size_t)(col0 + i * 16 + r) * K;
  }

  for (int k0 = 0; k0 < K; k0 += 32) {
    if (k0 + 512 < K) {  // stay ahead of the K loop (emits global_prefetch_b8)
      __builtin_prefetch(arow[0] + k0 + 512, 0, 1);
      __builtin_prefetch(arow[2] + k0 + 512, 0, 1);
      __builtin_prefetch(brow[0] + k0 + 512, 0, 1);
      __builtin_prefetch(brow[2] + k0 + 512, 0, 1);
    }
    Frag16 a[4];
    Frag16 b[4];
    const int ka = k0 + half * 8;
    const int kb = k0 + half * 16;
#pragma unroll
    for (int mi = 0; mi < 4; ++mi) {
      a[mi].h[0] = *reinterpret_cast<const bf16x8*>(arow[mi] + ka);
      a[mi].h[1] = *reinterpret_cast<const bf16x8*>(arow[mi] + ka + 16);
    }
#pragma unroll
    for (int ni = 0; ni < 4; ++ni) {
      b[ni].h[0] = *reinterpret_cast<const bf16x8*>(brow[ni] + ka);
      b[ni].h[1] = *reinterpret_cast<const bf16x8*>(brow[ni] + ka + 16);
    }

#pragma unroll
    for (int mi = 0; mi < 4; ++mi)
#pragma unroll
      for (int ni = 0; ni < 4; ++ni)
        acc[mi][ni] = __builtin_amdgcn_wmma_f32_16x16x32_bf16(
            false, a[mi].v, false, b[ni].v, (short)0, acc[mi][ni], false, false);
  }
}

__device__ __forceinline__ void zero_acc(f32x8 acc[4][4]) {
  const f32x8 z = {0.f, 0.f, 0.f, 0.f, 0.f, 0.f, 0.f, 0.f};
#pragma unroll
  for (int mi = 0; mi < 4; ++mi)
#pragma unroll
    for (int ni = 0; ni < 4; ++ni) acc[mi][ni] = z;
}

// ---------------------------------------------------------------------------
// QKV projections: 256 threads = 8 waves (4x2); block covers 256 x 128.
// blockIdx.z selects Q / K / V. V is written transposed (Vt[D][S]).
// ---------------------------------------------------------------------------
__global__ void __launch_bounds__(256)
qkv_gemm_kernel(const bf16_t* __restrict__ Xbf, const bf16_t* __restrict__ Wbf,
                const float* __restrict__ bq, const float* __restrict__ bk,
                const float* __restrict__ bv,
                bf16_t* __restrict__ Qbf, bf16_t* __restrict__ Kbf,
                bf16_t* __restrict__ Vtbf) {
  const int which = blockIdx.z;
  const bf16_t* W = Wbf + (size_t)which * D_DIM * D_DIM;
  const float* bias = (which == 0) ? bq : (which == 1) ? bk : bv;

  const int wid    = threadIdx.x >> 5;
  const int tile_m = blockIdx.y * 256 + (wid >> 1) * 64;
  const int tile_n = blockIdx.x * 128 + (wid & 1) * 64;

  f32x8 acc[4][4];
  zero_acc(acc);
  wmma_tile_64x64(Xbf, W, D_DIM, tile_m, tile_n, acc);

  const int lane = threadIdx.x & 31;
  const int half = lane >> 4;
  const int r    = lane & 15;
  bf16_t* QorK = (which == 0) ? Qbf : Kbf;
#pragma unroll
  for (int mi = 0; mi < 4; ++mi)
#pragma unroll
    for (int ni = 0; ni < 4; ++ni) {
      const int col = tile_n + ni * 16 + r;
      const float bb = bias[col];
#pragma unroll
      for (int j = 0; j < 8; ++j) {
        const int row = tile_m + mi * 16 + half * 8 + j;
        const float y = acc[mi][ni][j] + bb;
        if (which == 2)
          Vtbf[(size_t)col * S_DIM + row] = (bf16_t)y;
        else
          QorK[(size_t)row * D_DIM + col] = (bf16_t)y;
      }
    }
}

// ---------------------------------------------------------------------------
// Logits: QK[i,k] = (Q[i,:] . K[k,:]) / sqrt(D), fp32 out.
// ---------------------------------------------------------------------------
__global__ void __launch_bounds__(256)
qk_gemm_kernel(const bf16_t* __restrict__ Qbf, const bf16_t* __restrict__ Kbf,
               float* __restrict__ QK) {
  constexpr float kScale = 0.022097086912079608f;  // 1/sqrt(2048)
  const int wid    = threadIdx.x >> 5;
  const int tile_m = blockIdx.y * 256 + (wid >> 1) * 64;
  const int tile_n = blockIdx.x * 128 + (wid & 1) * 64;

  f32x8 acc[4][4];
  zero_acc(acc);
  wmma_tile_64x64(Qbf, Kbf, D_DIM, tile_m, tile_n, acc);

  const int lane = threadIdx.x & 31;
  const int half = lane >> 4;
  const int r    = lane & 15;
#pragma unroll
  for (int mi = 0; mi < 4; ++mi)
#pragma unroll
    for (int ni = 0; ni < 4; ++ni) {
      const int col = tile_n + ni * 16 + r;
#pragma unroll
      for (int j = 0; j < 8; ++j) {
        const int row = tile_m + mi * 16 + half * 8 + j;
        QK[(size_t)row * S_DIM + col] = acc[mi][ni][j] * kScale;
      }
    }
}

// ---------------------------------------------------------------------------
// Full-row softmax (denominator over ALL columns) then tril mask, bf16 out.
// One 256-thread block per row; LDS tree reductions.
// ---------------------------------------------------------------------------
__global__ void __launch_bounds__(256)
softmax_tril_kernel(const float* __restrict__ QK, bf16_t* __restrict__ P) {
  __shared__ float red[256];
  const int row = blockIdx.x;
  const int tid = threadIdx.x;
  const float* x = QK + (size_t)row * S_DIM;

  float lmax = -3.402823466e38f;
  for (int c = tid; c < S_DIM; c += 256) lmax = fmaxf(lmax, x[c]);
  red[tid] = lmax;
  __syncthreads();
  for (int s = 128; s > 0; s >>= 1) {
    if (tid < s) red[tid] = fmaxf(red[tid], red[tid + s]);
    __syncthreads();
  }
  const float m = red[0];
  __syncthreads();

  float lsum = 0.f;
  for (int c = tid; c < S_DIM; c += 256) lsum += __expf(x[c] - m);
  red[tid] = lsum;
  __syncthreads();
  for (int s = 128; s > 0; s >>= 1) {
    if (tid < s) red[tid] += red[tid + s];
    __syncthreads();
  }
  const float inv = 1.0f / red[0];

  for (int c = tid; c < S_DIM; c += 256) {
    const float p = __expf(x[c] - m) * inv;
    P[(size_t)row * S_DIM + c] = (c <= row) ? (bf16_t)p : (bf16_t)0.0f;
  }
}

// ---------------------------------------------------------------------------
// out[i,d] = sum_k P[i,k] * Vt[d,k]   (fp32 out)
// ---------------------------------------------------------------------------
__global__ void __launch_bounds__(256)
pv_gemm_kernel(const bf16_t* __restrict__ Pbf, const bf16_t* __restrict__ Vtbf,
               float* __restrict__ out) {
  const int wid    = threadIdx.x >> 5;
  const int tile_m = blockIdx.y * 256 + (wid >> 1) * 64;
  const int tile_n = blockIdx.x * 128 + (wid & 1) * 64;

  f32x8 acc[4][4];
  zero_acc(acc);
  wmma_tile_64x64(Pbf, Vtbf, S_DIM, tile_m, tile_n, acc);

  const int lane = threadIdx.x & 31;
  const int half = lane >> 4;
  const int r    = lane & 15;
#pragma unroll
  for (int mi = 0; mi < 4; ++mi)
#pragma unroll
    for (int ni = 0; ni < 4; ++ni) {
      const int col = tile_n + ni * 16 + r;
#pragma unroll
      for (int j = 0; j < 8; ++j) {
        const int row = tile_m + mi * 16 + half * 8 + j;
        out[(size_t)row * D_DIM + col] = acc[mi][ni][j];
      }
    }
}

// ---------------------------------------------------------------------------
// Launcher. Workspace layout (bytes), ~151 MB total:
//   Qbf  [0,            16.7M)   Kbf [16.7M, 33.5M)   Vtbf [33.5M, 50.3M)
//   QKf32[50.3M,       117.4M)   Pbf [117.4M, 151.0M)
//   Xbf/Wbf are lifetime-overlapped into the QKf32 region (used only before
//   the logits GEMM writes QK).
// ---------------------------------------------------------------------------
extern "C" void kernel_launch(void* const* d_in, const int* in_sizes, int n_in,
                              void* d_out, int out_size, void* d_ws, size_t ws_size,
                              hipStream_t stream) {
  const float* X  = (const float*)d_in[0];
  const float* Wq = (const float*)d_in[1];
  const float* bq = (const float*)d_in[2];
  const float* Wk = (const float*)d_in[3];
  const float* bk = (const float*)d_in[4];
  const float* Wv = (const float*)d_in[5];
  const float* bv = (const float*)d_in[6];
  float* out = (float*)d_out;

  const size_t SD2 = (size_t)S_DIM * D_DIM * sizeof(bf16_t);   // 16,777,216
  const size_t SS4 = (size_t)S_DIM * S_DIM * sizeof(float);    // 67,108,864

  char* ws = (char*)d_ws;
  bf16_t* Qbf  = (bf16_t*)(ws);
  bf16_t* Kbf  = (bf16_t*)(ws + SD2);
  bf16_t* Vtbf = (bf16_t*)(ws + 2 * SD2);
  float*  QKf  = (float*) (ws + 3 * SD2);
  bf16_t* Pbf  = (bf16_t*)(ws + 3 * SD2 + SS4);
  // Overlapped lifetimes (dead once QKf is written):
  bf16_t* Xbf  = (bf16_t*)(ws + 3 * SD2);
  bf16_t* Wbf  = (bf16_t*)(ws + 3 * SD2 + SD2);

  // 1) fp32 -> bf16 conversions
  {
    const int n4x = (S_DIM * D_DIM) / 4;
    cvt_f32_to_bf16<<<(n4x + 255) / 256, 256, 0, stream>>>(X, Xbf, n4x);
    const int n4w = (D_DIM * D_DIM) / 4;
    cvt_f32_to_bf16<<<(n4w + 255) / 256, 256, 0, stream>>>(Wq, Wbf, n4w);
    cvt_f32_to_bf16<<<(n4w + 255) / 256, 256, 0, stream>>>(
        Wk, Wbf + (size_t)D_DIM * D_DIM, n4w);
    cvt_f32_to_bf16<<<(n4w + 255) / 256, 256, 0, stream>>>(
        Wv, Wbf + 2 * (size_t)D_DIM * D_DIM, n4w);
  }

  // 2) Q/K/V projections (+bias); V stored transposed
  qkv_gemm_kernel<<<dim3(D_DIM / 128, S_DIM / 256, 3), 256, 0, stream>>>(
      Xbf, Wbf, bq, bk, bv, Qbf, Kbf, Vtbf);

  // 3) scaled logits QK = Q K^T / sqrt(D)
  qk_gemm_kernel<<<dim3(S_DIM / 128, S_DIM / 256), 256, 0, stream>>>(Qbf, Kbf, QKf);

  // 4) full-row softmax, then tril mask, bf16 P
  softmax_tril_kernel<<<S_DIM, 256, 0, stream>>>(QKf, Pbf);

  // 5) out = P @ V
  pv_gemm_kernel<<<dim3(D_DIM / 128, S_DIM / 256), 256, 0, stream>>>(Pbf, Vtbf, out);
}
